// StaticRwkvCore_47519518163482
// MI455X (gfx1250) — matrix operand order repaired and checked by reading, not compiled, not run on hardware
//
#include <hip/hip_runtime.h>
#include <math.h>

// ---------------- model constants ----------------
#define T_TOK   2048
#define DMODEL  1024
#define NLAYER  6
#define NHEAD   16
#define HEADK   64
#define FFN     3584
#define NVOCAB  50277

typedef __attribute__((ext_vector_type(16))) __bf16 v16bf;
typedef __attribute__((ext_vector_type(8)))  float  v8f;

// Let the backend pick native v_cvt_pk_bf16_f32 if available (RNE either way).
__device__ __forceinline__ __bf16 f2bf(float f) { return static_cast<__bf16>(f); }

__device__ __forceinline__ float sigmoidf_(float x) { return 1.f / (1.f + expf(-x)); }

// ---------------- LayerNorm (optionally gathering rows via tokens) ----------------
__global__ void ln_kernel(const float* __restrict__ src, const int* __restrict__ tokens,
                          const float* __restrict__ w, const float* __restrict__ b,
                          float* __restrict__ out) {
    const int t = blockIdx.x;
    const int tid = threadIdx.x;
    const int row = tokens ? tokens[t] : t;
    const float* xrow = src + (size_t)row * DMODEL;
    __shared__ float s1[256], s2[256];
    float vals[4];
    float sum = 0.f, sq = 0.f;
#pragma unroll
    for (int i = 0; i < 4; i++) {
        float v = xrow[tid + i * 256];
        vals[i] = v; sum += v; sq += v * v;
    }
    s1[tid] = sum; s2[tid] = sq;
    __syncthreads();
    for (int off = 128; off > 0; off >>= 1) {
        if (tid < off) { s1[tid] += s1[tid + off]; s2[tid] += s2[tid + off]; }
        __syncthreads();
    }
    const float mean = s1[0] * (1.f / DMODEL);
    const float var  = s2[0] * (1.f / DMODEL) - mean * mean;
    const float inv  = rsqrtf(var + 1e-5f);
#pragma unroll
    for (int i = 0; i < 4; i++) {
        int d = tid + i * 256;
        out[(size_t)t * DMODEL + d] = (vals[i] - mean) * inv * w[d] + b[d];
    }
}

// ---------------- token-shift mix: out = xn*m + shift(xn)*(1-m) ----------------
__global__ void mix_kernel(const float* __restrict__ xn, const float* __restrict__ m,
                           float* __restrict__ out) {
    const int t = blockIdx.x, tid = threadIdx.x;
#pragma unroll
    for (int i = 0; i < 4; i++) {
        int d = tid + i * 256;
        float cur  = xn[(size_t)t * DMODEL + d];
        float prev = (t > 0) ? xn[(size_t)(t - 1) * DMODEL + d] : 0.f;
        float mm = m[d];
        out[(size_t)t * DMODEL + d] = cur * mm + prev * (1.f - mm);
    }
}

// ---------------- WKV scan: one block (64 threads) per head ----------------
__global__ void wkv_kernel(const float* __restrict__ r, const float* __restrict__ k,
                           const float* __restrict__ v, const float* __restrict__ wp,
                           const float* __restrict__ ub, float* __restrict__ y) {
    const int h = blockIdx.x;
    const int j = threadIdx.x;
    __shared__ float sr[HEADK], sk[HEADK], sw[HEADK], su[HEADK];
    sw[j] = expf(-expf(wp[h * HEADK + j]));
    su[j] = ub[h * HEADK + j];
    float S[HEADK];
#pragma unroll
    for (int i = 0; i < HEADK; i++) S[i] = 0.f;
    __syncthreads();
    for (int t = 0; t < T_TOK; t++) {
        const size_t base = (size_t)t * DMODEL + h * HEADK;
        sr[j] = r[base + j];
        sk[j] = k[base + j];
        const float vj = v[base + j];
        __syncthreads();
        float acc = 0.f;
#pragma unroll
        for (int i = 0; i < HEADK; i++) {
            float a = sk[i] * vj;
            acc += sr[i] * (S[i] + su[i] * a);
            S[i] = sw[i] * S[i] + a;
        }
        y[base + j] = acc;
        __syncthreads();
    }
}

// ---------------- per-head GroupNorm (eps=64e-5) + silu(g) gate ----------------
__global__ void gn_gate_kernel(float* __restrict__ y, const float* __restrict__ gnw,
                               const float* __restrict__ gnb, const float* __restrict__ g) {
    const int wave = threadIdx.x >> 5, lane = threadIdx.x & 31;
    const int gid = blockIdx.x * 8 + wave;
    const int t = gid >> 4, h = gid & (NHEAD - 1);
    const size_t base = (size_t)t * DMODEL + h * HEADK;
    const int c0 = lane * 2, c1 = c0 + 1;
    const float y0 = y[base + c0], y1 = y[base + c1];
    float s = y0 + y1;
    float q = y0 * y0 + y1 * y1;
#pragma unroll
    for (int off = 16; off > 0; off >>= 1) {
        s += __shfl_xor(s, off);
        q += __shfl_xor(q, off);
    }
    const float mu  = s * (1.f / 64.f);
    const float var = q * (1.f / 64.f) - mu * mu;
    const float inv = rsqrtf(var + 64e-5f);
    const int d0 = h * HEADK + c0, d1 = d0 + 1;
    const float g0 = g[base + c0], g1 = g[base + c1];
    y[base + c0] = ((y0 - mu) * inv * gnw[d0] + gnb[d0]) * (g0 * sigmoidf_(g0));
    y[base + c1] = ((y1 - mu) * inv * gnw[d1] + gnb[d1]) * (g1 * sigmoidf_(g1));
}

// ---------------- elementwise epilogues ----------------
__global__ void sqrelu_kernel(float* __restrict__ p, int n) {
    int i = blockIdx.x * 256 + threadIdx.x;
    if (i < n) { float v = p[i]; v = v > 0.f ? v : 0.f; p[i] = v * v; }
}
__global__ void gate_add_kernel(float* __restrict__ x, const float* __restrict__ cr,
                                const float* __restrict__ cv, int n) {
    int i = blockIdx.x * 256 + threadIdx.x;
    if (i < n) x[i] += sigmoidf_(cr[i]) * cv[i];
}

// ---------------- bf16 WMMA GEMM (double-buffered) ----------------
// C[M,N] (+)= A[M,K] @ B[K,N], fp32 in HBM, bf16 WMMA with fp32 accumulate.
// Block tile 64(M) x 128(N); 8 waves as 4(M) x 2(N); wave = 1 A frag x 4 B frags
// -> 4 chained v_wmma_f32_16x16x32_bf16 per K-step.
// LDS is fragment-ordered [buf][group][lane][16 bf16] (32B contiguous per lane)
// and DOUBLE-BUFFERED: next tile's global loads issue before the barrier and are
// converted/stored into the other buffer after compute -> one barrier per K-step.
// Requires M % 64 == 0, K % 32 == 0 (true at all call sites); N arbitrary.

union BfPack4 { __bf16 p[4]; uint2 u; };

template <bool ADD>
__global__ __launch_bounds__(256) void gemm_bf16_kernel(const float* __restrict__ A,
                                                        const float* __restrict__ B,
                                                        float* __restrict__ C,
                                                        int M, int N, int K) {
    __shared__ __attribute__((aligned(32))) __bf16 ldsAf[2][4][32][16]; // 8 KB
    __shared__ __attribute__((aligned(32))) __bf16 ldsBf[2][8][32][16]; // 16 KB

    const int tid  = threadIdx.x;
    const int lane = tid & 31, wave = tid >> 5;
    const int wm = wave & 3, wn = wave >> 2;            // 4 x 2 wave grid
    const int tileM = blockIdx.y * 64, tileN = blockIdx.x * 128;

    // A loader: row = tid>>2 (0..63), kbase = (tid&3)*8 -> two float4 groups
    const int ar = tid >> 2;
    const int ac = (tid & 3) * 8;
    // B loader: kq = (tid>>5)*4 (0..28), cb = (tid&31)*4 (0..124)
    const int bk = (tid >> 5) * 4;
    const int bc = (tid & 31) * 4;
    const int gc = tileN + bc;

    float4 aReg0, aReg1;
    float  bReg[4][4];

    // ---- global tile load into registers ----
    auto loadGlobal = [&](int k0) {
        const float* Arow = A + (size_t)(tileM + ar) * K + k0 + ac;
        aReg0 = *reinterpret_cast<const float4*>(Arow);
        aReg1 = *reinterpret_cast<const float4*>(Arow + 4);
        const float* Bbase = B + (size_t)(k0 + bk) * N;
        if (gc + 4 <= N) {
#pragma unroll
            for (int i = 0; i < 4; i++)
                *reinterpret_cast<float4*>(bReg[i]) =
                    *reinterpret_cast<const float4*>(Bbase + (size_t)i * N + gc);
            if (k0 + 32 < K) __builtin_prefetch(Bbase + (size_t)32 * N + gc, 0, 1);
        } else {
#pragma unroll
            for (int i = 0; i < 4; i++)
#pragma unroll
                for (int j = 0; j < 4; j++) {
                    int cidx = gc + j;
                    int cl = cidx < N - 1 ? cidx : N - 1;   // clamp: branch-free load
                    float vv = Bbase[(size_t)i * N + cl];
                    bReg[i][j] = (cidx < N) ? vv : 0.f;
                }
        }
    };

    // ---- convert + store registers into fragment-ordered LDS buffer ----
    auto stageLDS = [&](int buf) {
        // A: two aligned groups of 4 K-values
#pragma unroll
        for (int gph = 0; gph < 2; gph++) {
            const int kbase = ac + gph * 4;
            const float4 v = gph ? aReg1 : aReg0;
            const int half = (kbase & 8) ? 1 : 0;
            const int la   = (ar & 15) + 16 * half;
            const int e0   = (kbase & 7) + ((kbase & 16) >> 1);
            BfPack4 pk;
            pk.p[0] = f2bf(v.x); pk.p[1] = f2bf(v.y);
            pk.p[2] = f2bf(v.z); pk.p[3] = f2bf(v.w);
            *reinterpret_cast<uint2*>(&ldsAf[buf][ar >> 4][la][e0]) = pk.u;
        }
        // B: 4 columns, each a 4-deep K run
        const int e0 = (bk & 7) + ((bk & 16) >> 1);
        const int lhalf = 16 * ((bk >> 3) & 1);
#pragma unroll
        for (int j = 0; j < 4; j++) {
            const int c = bc + j;
            BfPack4 pk;
            pk.p[0] = f2bf(bReg[0][j]); pk.p[1] = f2bf(bReg[1][j]);
            pk.p[2] = f2bf(bReg[2][j]); pk.p[3] = f2bf(bReg[3][j]);
            *reinterpret_cast<uint2*>(&ldsBf[buf][c >> 4][(c & 15) + lhalf][e0]) = pk.u;
        }
    };

    v8f acc[4] = {};

    // prologue: tile 0 into buffer 0
    loadGlobal(0);
    stageLDS(0);

    int cur = 0;
    for (int k0 = 0; k0 < K; k0 += 32) {
        const bool more = (k0 + 32) < K;
        if (more) loadGlobal(k0 + 32);        // overlap with compute below
        __syncthreads();                      // buffer `cur` now visible

        const v16bf af = *reinterpret_cast<const v16bf*>(ldsAf[cur][wm][lane]);
        v16bf bfr[4];
#pragma unroll
        for (int g = 0; g < 4; g++)
            bfr[g] = *reinterpret_cast<const v16bf*>(ldsBf[cur][wn * 4 + g][lane]);
#pragma unroll
        for (int g = 0; g < 4; g++)
            acc[g] = __builtin_amdgcn_wmma_f32_16x16x32_bf16(
                false, af, false, bfr[g], (short)0, acc[g], false, false);

        if (more) stageLDS(cur ^ 1);          // fill the other buffer
        cur ^= 1;
    }

    // ---- store C: VGPR j -> row = (lane>>4)*8 + j, col = lane&15 ----
    const int rbase = tileM + wm * 16 + (lane >> 4) * 8;
#pragma unroll
    for (int g = 0; g < 4; g++) {
        const int col = tileN + (wn * 4 + g) * 16 + (lane & 15);
        if (col < N) {
#pragma unroll
            for (int j = 0; j < 8; j++) {
                const size_t idx = (size_t)(rbase + j) * N + col;
                C[idx] = ADD ? (C[idx] + acc[g][j]) : acc[g][j];
            }
        }
    }
}

static inline dim3 gemm_grid(int M, int N) { return dim3((N + 127) / 128, (M + 63) / 64); }

// ---------------- host orchestration ----------------
extern "C" void kernel_launch(void* const* d_in, const int* in_sizes, int n_in,
                              void* d_out, int out_size, void* d_ws, size_t ws_size,
                              hipStream_t stream) {
    (void)in_sizes; (void)n_in; (void)out_size; (void)ws_size;

    const int*   tokens = (const int*)  d_in[0];
    const float* emb    = (const float*)d_in[1];
    const float* ln0w   = (const float*)d_in[2];
    const float* ln0b   = (const float*)d_in[3];
    const float* ln1w   = (const float*)d_in[4];
    const float* ln1b   = (const float*)d_in[5];
    const float* ln2w   = (const float*)d_in[6];
    const float* ln2b   = (const float*)d_in[7];
    const float* mixr   = (const float*)d_in[8];
    const float* mixk   = (const float*)d_in[9];
    const float* mixv   = (const float*)d_in[10];
    const float* mixg   = (const float*)d_in[11];
    const float* Wr     = (const float*)d_in[12];
    const float* Wk     = (const float*)d_in[13];
    const float* Wv     = (const float*)d_in[14];
    const float* Wg     = (const float*)d_in[15];
    const float* Wo     = (const float*)d_in[16];
    const float* wp     = (const float*)d_in[17];
    const float* ub     = (const float*)d_in[18];
    const float* gnw    = (const float*)d_in[19];
    const float* gnb    = (const float*)d_in[20];
    const float* cmk    = (const float*)d_in[21];
    const float* cmr    = (const float*)d_in[22];
    const float* Wck    = (const float*)d_in[23];
    const float* Wcv    = (const float*)d_in[24];
    const float* Wcr    = (const float*)d_in[25];
    const float* lnow   = (const float*)d_in[26];
    const float* lnob   = (const float*)d_in[27];
    const float* head   = (const float*)d_in[28];

    const size_t TD = (size_t)T_TOK * DMODEL;
    const size_t TF = (size_t)T_TOK * FFN;
    const size_t DD = (size_t)DMODEL * DMODEL;
    const size_t DF = (size_t)DMODEL * FFN;

    float* ws  = (float*)d_ws;
    float* x   = ws;            // [T,D] residual stream
    float* xn  = x  + TD;       // [T,D] LN output
    float* tmp = xn + TD;       // [T,D] mixed GEMM input
    float* rb  = tmp + TD;      // [T,D] r / cmix-r
    float* kb  = rb + TD;       // [T,D] k / cmix-v out
    float* vb  = kb + TD;       // [T,D] v
    float* gb  = vb + TD;       // [T,D] g
    float* yb  = gb + TD;       // [T,D] wkv out
    float* kkb = yb + TD;       // [T,F] channel-mix hidden

    const dim3 blk(256);

    // embed + ln0
    ln_kernel<<<T_TOK, blk, 0, stream>>>(emb, tokens, ln0w, ln0b, x);

    for (int l = 0; l < NLAYER; l++) {
        const size_t oD = (size_t)l * DMODEL;
        // ---- time mixing ----
        ln_kernel<<<T_TOK, blk, 0, stream>>>(x, nullptr, ln1w + oD, ln1b + oD, xn);

        mix_kernel<<<T_TOK, blk, 0, stream>>>(xn, mixr + oD, tmp);
        gemm_bf16_kernel<false><<<gemm_grid(T_TOK, DMODEL), blk, 0, stream>>>(
            tmp, Wr + l * DD, rb, T_TOK, DMODEL, DMODEL);
        mix_kernel<<<T_TOK, blk, 0, stream>>>(xn, mixk + oD, tmp);
        gemm_bf16_kernel<false><<<gemm_grid(T_TOK, DMODEL), blk, 0, stream>>>(
            tmp, Wk + l * DD, kb, T_TOK, DMODEL, DMODEL);
        mix_kernel<<<T_TOK, blk, 0, stream>>>(xn, mixv + oD, tmp);
        gemm_bf16_kernel<false><<<gemm_grid(T_TOK, DMODEL), blk, 0, stream>>>(
            tmp, Wv + l * DD, vb, T_TOK, DMODEL, DMODEL);
        mix_kernel<<<T_TOK, blk, 0, stream>>>(xn, mixg + oD, tmp);
        gemm_bf16_kernel<false><<<gemm_grid(T_TOK, DMODEL), blk, 0, stream>>>(
            tmp, Wg + l * DD, gb, T_TOK, DMODEL, DMODEL);

        wkv_kernel<<<NHEAD, 64, 0, stream>>>(rb, kb, vb, wp + oD, ub + oD, yb);
        gn_gate_kernel<<<(T_TOK * NHEAD) / 8, blk, 0, stream>>>(yb, gnw + oD, gnb + oD, gb);
        gemm_bf16_kernel<true><<<gemm_grid(T_TOK, DMODEL), blk, 0, stream>>>(
            yb, Wo + l * DD, x, T_TOK, DMODEL, DMODEL);   // x += y @ Wo

        // ---- channel mixing ----
        ln_kernel<<<T_TOK, blk, 0, stream>>>(x, nullptr, ln2w + oD, ln2b + oD, xn);

        mix_kernel<<<T_TOK, blk, 0, stream>>>(xn, cmk + oD, tmp);
        gemm_bf16_kernel<false><<<gemm_grid(T_TOK, FFN), blk, 0, stream>>>(
            tmp, Wck + l * DF, kkb, T_TOK, FFN, DMODEL);
        sqrelu_kernel<<<(int)((TF + 255) / 256), blk, 0, stream>>>(kkb, (int)TF);

        mix_kernel<<<T_TOK, blk, 0, stream>>>(xn, cmr + oD, tmp);
        gemm_bf16_kernel<false><<<gemm_grid(T_TOK, DMODEL), blk, 0, stream>>>(
            tmp, Wcr + l * DD, rb, T_TOK, DMODEL, DMODEL);

        gemm_bf16_kernel<false><<<gemm_grid(T_TOK, DMODEL), blk, 0, stream>>>(
            kkb, Wcv + l * DF, kb, T_TOK, DMODEL, FFN);

        gate_add_kernel<<<(int)((TD + 255) / 256), blk, 0, stream>>>(x, rb, kb, (int)TD);
    }

    // final LN + head projection (logits straight into d_out, fp32)
    ln_kernel<<<T_TOK, blk, 0, stream>>>(x, nullptr, lnow, lnob, xn);
    gemm_bf16_kernel<false><<<gemm_grid(T_TOK, NVOCAB), blk, 0, stream>>>(
        xn, head, (float*)d_out, T_TOK, NVOCAB, DMODEL);
}
